// pointnet2_seg_msg_radar_2430951489690
// MI455X (gfx1250) — compile-verified, matrix operand order checked by reading, not run on hardware
//
#include <hip/hip_runtime.h>
#include <hip/hip_bf16.h>

typedef _Float16 f16;
typedef __attribute__((ext_vector_type(8)))  _Float16 v8h;
typedef __attribute__((ext_vector_type(16))) _Float16 v16h;
typedef __attribute__((ext_vector_type(8)))  float    v8f;

#define NPT   32768
#define NROWS 131072      // B*N = 4*32768
#define EPSF  1e-5f

// ---------------------------------------------------------------------------
// Front network: everything before fp3 collapses to 4 samples (1 per batch).
// ---------------------------------------------------------------------------
struct FrontW {
    const float* W[18];  const float* G[18];  const float* Bv[18];   // 6 branches x 3 layers
    const float* fpW[4]; const float* fpG[4]; const float* fpB[4];   // fp1(2) + fp2(2)
    const float* W0;                                                 // fp3 layer0, 135x128
};

__device__ __forceinline__ void fc_bn(const float* in, int nin, float* out, int nout,
                                      const float* W, const float* g, const float* bb, int tid) {
    for (int c = tid; c < nout; c += 256) {
        float y[4];
        for (int s = 0; s < 4; ++s) {
            float acc = 0.f;
            const float* row = in + s * 512;
            for (int j = 0; j < nin; ++j) acc += row[j] * W[j * nout + c];
            y[s] = acc;
        }
        float mu = 0.25f * (y[0] + y[1] + y[2] + y[3]);
        float var = 0.f;
        for (int s = 0; s < 4; ++s) { float d = y[s] - mu; var += d * d; }
        var *= 0.25f;
        float rs = rsqrtf(var + EPSF);
        float gc = g[c], bc = bb[c];
        for (int s = 0; s < 4; ++s) {
            float t = (y[s] - mu) * rs * gc + bc;
            out[s * 512 + c] = t > 0.f ? t : 0.f;
        }
    }
    __syncthreads();
}

__device__ __forceinline__ void setcat3(float* cat, const float* feat, int nfeat, int tid) {
    int n = nfeat + 3;
    for (int i = tid; i < 4 * n; i += 256) {
        int s = i / n, c = i % n;
        cat[s * 512 + c] = (c < 3) ? 0.f : feat[s * 512 + (c - 3)];
    }
    __syncthreads();
}

__device__ __forceinline__ void cat2(float* cat, const float* a, int na,
                                     const float* b, int nb, int tid) {
    int n = na + nb;
    for (int i = tid; i < 4 * n; i += 256) {
        int s = i / n, c = i % n;
        cat[s * 512 + c] = (c < na) ? a[s * 512 + c] : b[s * 512 + (c - na)];
    }
    __syncthreads();
}

__device__ __forceinline__ void copyoff(float* dst, int off, const float* src, int n, int tid) {
    for (int i = tid; i < 4 * n; i += 256) {
        int s = i / n, c = i % n;
        dst[s * 512 + off + c] = src[s * 512 + c];
    }
    __syncthreads();
}

__device__ __forceinline__ void branch3(const FrontW& fw, int br, const float* in, int nin,
                                        int n0, int n1, int n2,
                                        float* tA, float* tB, float* outbuf, int outoff, int tid) {
    int l = br * 3;
    fc_bn(in, nin, tA, n0, fw.W[l + 0], fw.G[l + 0], fw.Bv[l + 0], tid);
    fc_bn(tA, n0, tB, n1, fw.W[l + 1], fw.G[l + 1], fw.Bv[l + 1], tid);
    fc_bn(tB, n1, tA, n2, fw.W[l + 2], fw.G[l + 2], fw.Bv[l + 2], tid);
    copyoff(outbuf, outoff, tA, n2, tid);
}

__global__ __launch_bounds__(256) void front_kernel(FrontW fw, const float* __restrict__ pts,
                                                    float* __restrict__ v_out,
                                                    float* __restrict__ base_out,
                                                    float* __restrict__ zero_ptr, int nzero) {
    __shared__ float bufA[4 * 512];
    __shared__ float bufB[4 * 512];
    __shared__ float cat[4 * 512];
    __shared__ float L1[4 * 512];
    __shared__ float L2b[4 * 512];
    __shared__ float L3b[4 * 512];
    const int tid = threadIdx.x;

    for (int i = tid; i < nzero; i += 256) zero_ptr[i] = 0.f;        // stats accumulators

    // sa1 input: [0,0,0, l0_points[b,0,0:4]]  (gx == 0 always)
    for (int i = tid; i < 4 * 7; i += 256) {
        int s = i / 7, c = i % 7;
        cat[s * 512 + c] = (c < 3) ? 0.f : pts[s * NPT * 4 + (c - 3)];
    }
    __syncthreads();

    branch3(fw, 0, cat, 7, 32, 32, 64,  bufA, bufB, L1, 0,  tid);
    branch3(fw, 1, cat, 7, 64, 64, 128, bufA, bufB, L1, 64, tid);
    setcat3(cat, L1, 192, tid);
    branch3(fw, 2, cat, 195, 32, 32, 64,  bufA, bufB, L2b, 0,  tid);
    branch3(fw, 3, cat, 195, 64, 64, 128, bufA, bufB, L2b, 64, tid);
    setcat3(cat, L2b, 192, tid);
    branch3(fw, 4, cat, 195, 64, 64, 128, bufA, bufB, L3b, 0,   tid);
    branch3(fw, 5, cat, 195, 64, 64, 128, bufA, bufB, L3b, 128, tid);

    cat2(cat, L3b, 256, L2b, 192, tid);                               // fp1 input: 448
    fc_bn(cat, 448, bufA, 256, fw.fpW[0], fw.fpG[0], fw.fpB[0], tid);
    fc_bn(bufA, 256, bufB, 256, fw.fpW[1], fw.fpG[1], fw.fpB[1], tid);  // l2' in bufB
    cat2(cat, bufB, 256, L1, 192, tid);                               // fp2 input: 448
    fc_bn(cat, 448, bufA, 128, fw.fpW[2], fw.fpG[2], fw.fpB[2], tid);
    fc_bn(bufA, 128, bufB, 128, fw.fpW[3], fw.fpG[3], fw.fpB[3], tid);  // v in bufB

    for (int i = tid; i < 4 * 128; i += 256) {
        int s = i / 128, c = i % 128;
        v_out[i] = bufB[s * 512 + c];
    }
    __syncthreads();
    // base[b][n] = v[b] . W0[0:128, n]   (constant-channel part of fp3 layer 0)
    for (int i = tid; i < 4 * 128; i += 256) {
        int s = i / 128, n = i % 128;
        float acc = 0.f;
        for (int j = 0; j < 128; ++j) acc += bufB[s * 512 + j] * fw.W0[j * 128 + n];
        base_out[i] = acc;
    }
}

// ---------------------------------------------------------------------------
// Weight convert: fp32 row-major (K x N) -> f16, pre-swizzled 32x16 B-fragments.
// Per tile (kt,nt): 512 halves; lane*16 contiguous halves per lane.
// ---------------------------------------------------------------------------
__global__ void cvt_kernel(const float* __restrict__ src, int srcK, int srcN,
                           int Kp, int Np, f16* __restrict__ dst) {
    int total = Kp * Np;
    int NT = Np >> 4;
    for (int idx = blockIdx.x * blockDim.x + threadIdx.x; idx < total;
         idx += gridDim.x * blockDim.x) {
        int t    = idx >> 9;
        int sl   = idx & 511;
        int lane = sl >> 4;
        int s    = sl & 15;
        int kt = t / NT, nt = t % NT;
        int vg = s >> 1, p = s & 1, hi = lane >> 4;
        int klocal = (vg < 4 ? vg * 2 + p : 16 + (vg - 4) * 2 + p) + hi * 8;
        int k = kt * 32 + klocal;
        int n = nt * 16 + (lane & 15);
        float val = (k < srcK && n < srcN) ? src[k * srcN + n] : 0.f;
        dst[idx] = (f16)val;
    }
}

// ---------------------------------------------------------------------------
// BN-fold: (sum, sumsq, gamma, beta) -> (scale, shift)
// ---------------------------------------------------------------------------
__global__ void stats_kernel(const float* __restrict__ sum, const float* __restrict__ sumsq,
                             const float* __restrict__ g, const float* __restrict__ b,
                             float* __restrict__ scale, float* __restrict__ shift,
                             int C, float invN) {
    int i = blockIdx.x * blockDim.x + threadIdx.x;
    if (i < C) {
        float mu  = sum[i] * invN;
        float var = sumsq[i] * invN - mu * mu;
        float rs  = rsqrtf(var + EPSF);
        float sc  = g[i] * rs;
        scale[i] = sc;
        shift[i] = b[i] - mu * sc;
    }
}

// ---------------------------------------------------------------------------
// WMMA GEMM pass: 256 threads = 8 waves, 128 rows/block.
//  A-fragments (16x32 f16 per wave) are loaded ONCE into registers and reused
//  across all output-column tiles; B-fragments stream from the pre-swizzled
//  f16 weight buffer (L2-resident, 32 contiguous bytes per lane).
//  IN_L0 : build X from (l0_points, l0_xyz), acc init = base[b][n]
//  !IN_L0: X = relu(Yprev*scale + shift) from f16 buffer (scale/shift in LDS)
//  OUT_CLS: add bias, write fp32 transposed (B,6,N); else store f16 Y + stats
// ---------------------------------------------------------------------------
template<int CIN, int COUT, bool IN_L0, bool OUT_CLS>
__global__ __launch_bounds__(256) void gemm_kernel(
    const f16*  __restrict__ Xin,
    const float* __restrict__ scale, const float* __restrict__ shift,
    const float* __restrict__ pts,   const float* __restrict__ xyz,
    const float* __restrict__ basev,
    const f16*  __restrict__ Wh,
    f16*  __restrict__ Yout,
    float* __restrict__ gSum, float* __restrict__ gSumsq,
    const float* __restrict__ clsBias, float* __restrict__ out)
{
    extern __shared__ char smem[];
    f16*   Xs       = (f16*)smem;
    float* ldsSum   = (float*)(smem + 128 * CIN * sizeof(f16));
    float* ldsSumsq = ldsSum + COUT;
    float* ldsScale = ldsSumsq + COUT;   // CIN entries (unused for IN_L0)
    float* ldsShift = ldsScale + CIN;

    const int tid = threadIdx.x;
    const int blockRow = blockIdx.x * 128;

    if constexpr (!OUT_CLS) {
        for (int i = tid; i < 2 * COUT; i += 256) ldsSum[i] = 0.f;
    }
    if constexpr (!IN_L0) {
        for (int i = tid; i < CIN; i += 256) {
            ldsScale[i] = scale[i];
            ldsShift[i] = shift[i];
        }
    }
    __syncthreads();

    // ---- stage normalized X tile into LDS (f16) ----
    if constexpr (IN_L0) {
        for (int i = tid; i < 128 * CIN; i += 256) {
            int r = i / CIN, c = i % CIN;
            int R = blockRow + r;
            int b = R >> 15, n = R & (NPT - 1);
            float val = 0.f;
            if (c < 4)      val = pts[(b * NPT + n) * 4 + c];
            else if (c < 7) val = xyz[(b * NPT + n) * 3 + (c - 4)];
            Xs[i] = (f16)val;
        }
    } else {
        for (int i = tid; i < 128 * CIN; i += 256) {
            int r = i / CIN, c = i % CIN;
            int R = blockRow + r;
            float y = (float)Xin[R * CIN + c];
            y = y * ldsScale[c] + ldsShift[c];
            Xs[i] = (f16)(y > 0.f ? y : 0.f);
        }
    }
    __syncthreads();

    const int wave = tid >> 5;
    const int lane = tid & 31;
    const int col  = lane & 15;
    const int hi   = lane >> 4;
    const int waveRow = wave * 16;
    const int bIdx = blockRow >> 15;

    constexpr int NT = COUT / 16;
    constexpr int KT = CIN / 32;

    // ---- load all A fragments once (invariant across nt) ----
    v16h afrag[KT];
#pragma unroll
    for (int kt = 0; kt < KT; ++kt) {
        const f16* ap = Xs + (waveRow + col) * CIN + kt * 32 + hi * 8;
        v8h alo = *(const v8h*)ap;
        v8h ahi = *(const v8h*)(ap + 16);
        afrag[kt] = __builtin_shufflevector(alo, ahi,
            0, 1, 2, 3, 4, 5, 6, 7, 8, 9, 10, 11, 12, 13, 14, 15);
    }

#pragma unroll 1
    for (int nt = 0; nt < NT; ++nt) {
        v8f acc;
        if constexpr (IN_L0) {
            float bv = basev[bIdx * 128 + nt * 16 + col];
#pragma unroll
            for (int r = 0; r < 8; ++r) acc[r] = bv;
        } else {
#pragma unroll
            for (int r = 0; r < 8; ++r) acc[r] = 0.f;
        }

#pragma unroll
        for (int kt = 0; kt < KT; ++kt) {
            // B fragment: pre-swizzled, 16 contiguous halves per lane
            const f16* bp = Wh + ((kt * NT + nt) << 9) + lane * 16;
            v8h blo = *(const v8h*)bp;
            v8h bhi = *(const v8h*)(bp + 8);
            v16h bf = __builtin_shufflevector(blo, bhi,
                0, 1, 2, 3, 4, 5, 6, 7, 8, 9, 10, 11, 12, 13, 14, 15);
            acc = __builtin_amdgcn_wmma_f32_16x16x32_f16(
                false, afrag[kt], false, bf, (short)0, acc, false, false);
        }

        if constexpr (OUT_CLS) {
            if (col < 6) {
                float bias = clsBias[col];
#pragma unroll
                for (int r = 0; r < 8; ++r) {
                    int R = blockRow + waveRow + hi * 8 + r;
                    int b = R >> 15, n = R & (NPT - 1);
                    out[(b * 6 + col) * NPT + n] = acc[r] + bias;
                }
            }
        } else {
            float s = 0.f, s2 = 0.f;
            int nglob = nt * 16 + col;
#pragma unroll
            for (int r = 0; r < 8; ++r) {
                float yv = acc[r];
                s += yv; s2 += yv * yv;
                int R = blockRow + waveRow + hi * 8 + r;
                Yout[R * COUT + nglob] = (f16)yv;
            }
            atomicAdd(&ldsSum[nglob], s);
            atomicAdd(&ldsSumsq[nglob], s2);
        }
    }

    if constexpr (!OUT_CLS) {
        __syncthreads();
        for (int i = tid; i < COUT; i += 256) {
            atomicAdd(&gSum[i],   ldsSum[i]);
            atomicAdd(&gSumsq[i], ldsSumsq[i]);
        }
    }
}

static inline size_t gemm_smem(int cin, int cout) {
    return (size_t)128 * cin * 2 + (size_t)2 * cout * 4 + (size_t)2 * cin * 4;
}

// ---------------------------------------------------------------------------
// Host launch
// ---------------------------------------------------------------------------
extern "C" void kernel_launch(void* const* d_in, const int* in_sizes, int n_in,
                              void* d_out, int out_size, void* d_ws, size_t ws_size,
                              hipStream_t stream) {
    auto F = [&](int i) { return (const float*)d_in[i]; };
    const float* xyz = F(0);
    const float* pts = F(1);

    // workspace carve-up
    char* ws = (char*)d_ws;
    size_t off = 0;
    auto alloc = [&](size_t bytes) {
        char* p = ws + off;
        off = (off + bytes + 255) & ~(size_t)255;
        return p;
    };
    float* v_buf    = (float*)alloc(4 * 128 * 4);
    float* base_buf = (float*)alloc(4 * 128 * 4);
    float* stats    = (float*)alloc(5 * 2 * 256 * 4);  // [layer][sum|sumsq][256]
    float* ssb      = (float*)alloc(5 * 2 * 256 * 4);  // [layer][scale|shift][256]
    f16* Wh0 = (f16*)alloc((size_t)32  * 128 * 2);
    f16* Wh1 = (f16*)alloc((size_t)128 * 128 * 2);
    f16* Wh2 = (f16*)alloc((size_t)128 * 128 * 2);
    f16* Wh3 = (f16*)alloc((size_t)128 * 256 * 2);
    f16* Wh4 = (f16*)alloc((size_t)256 * 128 * 2);
    f16* Wh5 = (f16*)alloc((size_t)128 * 16 * 2);
    f16* bufA = (f16*)alloc((size_t)NROWS * 128 * 2);
    f16* bufB = (f16*)alloc((size_t)NROWS * 256 * 2);

    // weights -> f16 swizzled fragments
    cvt_kernel<<<16,  256, 0, stream>>>(F(68) + 128 * 128, 7,   128, 32,  128, Wh0); // fp3 L0 rows 128..134
    cvt_kernel<<<64,  256, 0, stream>>>(F(71), 128, 128, 128, 128, Wh1);             // fp3 L1
    cvt_kernel<<<64,  256, 0, stream>>>(F(74), 128, 128, 128, 128, Wh2);             // fp3 L2
    cvt_kernel<<<128, 256, 0, stream>>>(F(77), 128, 256, 128, 256, Wh3);             // head L0
    cvt_kernel<<<128, 256, 0, stream>>>(F(80), 256, 128, 256, 128, Wh4);             // head L1
    cvt_kernel<<<8,   256, 0, stream>>>(F(83), 128, 6,   128, 16,  Wh5);             // cls

    // front network (leaf order: l0_xyz, l0_points, then params in insertion order)
    FrontW fw;
    const int brBase[6] = {2, 11, 20, 29, 38, 47};
    for (int br = 0; br < 6; ++br)
        for (int l = 0; l < 3; ++l) {
            int idx = brBase[br] + l * 3;
            fw.W[br * 3 + l]  = F(idx);
            fw.G[br * 3 + l]  = F(idx + 1);
            fw.Bv[br * 3 + l] = F(idx + 2);
        }
    const int fpBase[4] = {56, 59, 62, 65};
    for (int i = 0; i < 4; ++i) {
        fw.fpW[i] = F(fpBase[i]);
        fw.fpG[i] = F(fpBase[i] + 1);
        fw.fpB[i] = F(fpBase[i] + 2);
    }
    fw.W0 = F(68);
    front_kernel<<<1, 256, 0, stream>>>(fw, pts, v_buf, base_buf, stats, 5 * 2 * 256);

    const float invN = 1.f / (float)NROWS;
    const int GRID = NROWS / 128;
    float* out = (float*)d_out;

    // L0: fp3 layer0 (K=32 effective) -> bufA, stats[0]
    gemm_kernel<32, 128, true, false><<<GRID, 256, gemm_smem(32, 128), stream>>>(
        nullptr, nullptr, nullptr, pts, xyz, base_buf, Wh0,
        bufA, stats + 0, stats + 256, nullptr, nullptr);
    stats_kernel<<<1, 256, 0, stream>>>(stats + 0, stats + 256, F(69), F(70),
                                        ssb + 0, ssb + 256, 128, invN);
    // L1: fp3 layer1
    gemm_kernel<128, 128, false, false><<<GRID, 256, gemm_smem(128, 128), stream>>>(
        bufA, ssb + 0, ssb + 256, nullptr, nullptr, nullptr, Wh1,
        bufB, stats + 512, stats + 768, nullptr, nullptr);
    stats_kernel<<<1, 256, 0, stream>>>(stats + 512, stats + 768, F(72), F(73),
                                        ssb + 512, ssb + 768, 128, invN);
    // L2: fp3 layer2
    gemm_kernel<128, 128, false, false><<<GRID, 256, gemm_smem(128, 128), stream>>>(
        bufB, ssb + 512, ssb + 768, nullptr, nullptr, nullptr, Wh2,
        bufA, stats + 1024, stats + 1280, nullptr, nullptr);
    stats_kernel<<<1, 256, 0, stream>>>(stats + 1024, stats + 1280, F(75), F(76),
                                        ssb + 1024, ssb + 1280, 128, invN);
    // L3: head layer0 (128 -> 256)
    gemm_kernel<128, 256, false, false><<<GRID, 256, gemm_smem(128, 256), stream>>>(
        bufA, ssb + 1024, ssb + 1280, nullptr, nullptr, nullptr, Wh3,
        bufB, stats + 1536, stats + 1792, nullptr, nullptr);
    stats_kernel<<<1, 256, 0, stream>>>(stats + 1536, stats + 1792, F(78), F(79),
                                        ssb + 1536, ssb + 1792, 256, invN);
    // L4: head layer1 (256 -> 128)
    gemm_kernel<256, 128, false, false><<<GRID, 256, gemm_smem(256, 128), stream>>>(
        bufB, ssb + 1536, ssb + 1792, nullptr, nullptr, nullptr, Wh4,
        bufA, stats + 2048, stats + 2304, nullptr, nullptr);
    stats_kernel<<<1, 256, 0, stream>>>(stats + 2048, stats + 2304, F(81), F(82),
                                        ssb + 2048, ssb + 2304, 128, invN);
    // L5: cls (128 -> 6, padded to 16), fp32 transposed output + bias
    gemm_kernel<128, 16, false, true><<<GRID, 256, gemm_smem(128, 16), stream>>>(
        bufA, ssb + 2048, ssb + 2304, nullptr, nullptr, nullptr, Wh5,
        nullptr, nullptr, nullptr, F(84), out);
}
